// Codebook_66949950210646
// MI455X (gfx1250) — matrix operand order repaired and checked by reading, not compiled, not run on hardware
//
#include <hip/hip_runtime.h>

// ---------------------------------------------------------------------------
// VQ bottleneck (down-proj -> codebook argmin -> gather -> up-proj + losses)
// MI455X / gfx1250: wave32, bf16 WMMA 16x16x32, fp32 accumulate.
// Roofline: ~336 MB HBM traffic (14.4 us) vs 68.7 GFLOP bf16 (~13.7 us) ->
// balanced; single fused pass over tokens, weights/codebook L2-resident.
// ---------------------------------------------------------------------------

typedef __bf16  v16bf __attribute__((ext_vector_type(16)));
typedef float   v8f   __attribute__((ext_vector_type(8)));

constexpr int BB   = 16;
constexpr int TT   = 4096;
constexpr int DIN  = 512;
constexpr int DEMB = 256;
constexpr int KCB  = 1024;
constexpr int BT   = BB * TT;          // 65536 tokens
constexpr int TILE_M = 128;            // tokens per workgroup
constexpr int NTILES = BT / TILE_M;    // 512 workgroups

// d_out layout (floats, concatenated in reference return order)
constexpr size_t O_ZQ     = 0;                                  // [BT, 512]
constexpr size_t O_ZED    = (size_t)BT * DIN;                   // [BT, 256]
constexpr size_t O_CODE   = O_ZED + (size_t)BT * DEMB;          // [BT]
constexpr size_t O_CLOSS  = O_CODE + BT;                        // [16]
constexpr size_t O_CBLOSS = O_CLOSS + BB;                       // [16]

static __device__ __forceinline__ v8f wmma_bf16(v16bf a, v16bf b, v8f c) {
    // D = A(16x32 bf16) x B(32x16 bf16) + C(16x16 f32)
    return __builtin_amdgcn_wmma_f32_16x16x32_bf16(
        /*neg_a=*/false, a, /*neg_b=*/false, b,
        /*c_mod=*/(short)0, c, /*reuse_a=*/false, /*reuse_b=*/false);
}

// Build a 16-bit A fragment (16x32 slice) per the ISA A-layout:
// lanes 0-15 : M=lane,    K = {kb+0..7, kb+16..23}
// lanes16-31 : M=lane-16, K = {kb+8..15, kb+24..31}
static __device__ __forceinline__ v16bf load_a_f32(const float* row, int kb, int sub) {
    const int k1 = kb + sub * 8;
    const int k2 = kb + 16 + sub * 8;
    v16bf a;
#pragma unroll
    for (int j = 0; j < 8; ++j) a[j]     = (__bf16)row[k1 + j];
#pragma unroll
    for (int j = 0; j < 8; ++j) a[8 + j] = (__bf16)row[k2 + j];
    return a;
}

static __device__ __forceinline__ v16bf load_a_bf16(const __bf16* row, int kb, int sub) {
    const int k1 = kb + sub * 8;
    const int k2 = kb + 16 + sub * 8;
    v16bf a;
#pragma unroll
    for (int j = 0; j < 8; ++j) a[j]     = row[k1 + j];
#pragma unroll
    for (int j = 0; j < 8; ++j) a[8 + j] = row[k2 + j];
    return a;
}

// ---------------------------------------------------------------------------
// Prep: convert weights to bf16, layout unchanged (L2-resident operands for WMMA B)
// ---------------------------------------------------------------------------
__global__ void vq_prep_weights(const float* __restrict__ Wd,   // [256,512]
                                const float* __restrict__ Wu,   // [512,256]
                                void* wdt_, void* wut_) {
    __bf16* wdt = (__bf16*)wdt_;   // [256,512]  wdt[e*512+d] = Wd[e,d]
    __bf16* wut = (__bf16*)wut_;   // [512,256]  wut[d*256+e] = Wu[d,e]
    int i = blockIdx.x * 256 + threadIdx.x;      // 0 .. 131071
    {
        int d = i >> 8, e = i & 255;
        wdt[e * DIN + d] = (__bf16)Wd[e * DIN + d];
    }
    {
        int e = i >> 9, d = i & 511;
        wut[d * DEMB + e] = (__bf16)Wu[d * DEMB + e];
    }
}

// Prep: codebook bf16 (row + transposed) and squared norms
__global__ void vq_prep_codebook(const float* __restrict__ cb,  // [1024,256]
                                 void* cbr_, void* cbt_,
                                 float* __restrict__ cnorm) {
    __bf16* cbr = (__bf16*)cbr_;   // [1024,256]
    __bf16* cbt = (__bf16*)cbt_;   // [256,1024]
    int k = blockIdx.x, e = threadIdx.x;
    float v = cb[k * DEMB + e];
    cbr[k * DEMB + e] = (__bf16)v;
    cbt[e * KCB + k]  = (__bf16)v;
    __shared__ float red[256];
    red[e] = v * v;
    __syncthreads();
#pragma unroll
    for (int s = 128; s > 0; s >>= 1) {
        if (e < s) red[e] += red[e + s];
        __syncthreads();
    }
    if (e == 0) cnorm[k] = red[0];
}

// ---------------------------------------------------------------------------
// Fused main kernel: 8 waves, each wave owns a 16-token strip of a 128-token
// tile. GEMM1 (down-proj) -> LDS exchange -> GEMM2+argmin -> gather ->
// loss -> GEMM3 (up-proj).
// ---------------------------------------------------------------------------
__global__ __launch_bounds__(256)
void vq_main(const float* __restrict__ ze,     // [BT,512]
             const void* wdt_, const void* wut_,
             const void* cbt_, const void* cbr_,
             const float* __restrict__ cnorm,  // [1024]
             float* __restrict__ lpart,        // [512] per-tile sq-err sums
             float* __restrict__ out) {
    const __bf16* wdt = (const __bf16*)wdt_;
    const __bf16* wut = (const __bf16*)wut_;
    const __bf16* cbt = (const __bf16*)cbt_;
    const __bf16* cbr = (const __bf16*)cbr_;

    __shared__ __bf16 lds_zd[8 * 16 * DEMB];   // 64 KB: bf16 z_e_down tile
    __shared__ int    lds_code[TILE_M];
    __shared__ float  lds_loss[8];

    const int wave = threadIdx.x >> 5;
    const int lane = threadIdx.x & 31;
    const int sub  = lane >> 4;        // lane half (A/C layout split)
    const int lcol = lane & 15;
    const int rowg = blockIdx.x * TILE_M + wave * 16;  // wave's first token

    // ---------------- GEMM1: z_down[16,256] = z_e[16,512] x Wd^T ----------
    v8f acc[16];
#pragma unroll
    for (int nt = 0; nt < 16; ++nt) acc[nt] = {};

    const float* arow = ze + (size_t)(rowg + lcol) * DIN;
    for (int kc = 0; kc < 16; ++kc) {               // K = 512 in chunks of 32
        const int kb = kc * 32;
        const v16bf a = load_a_f32(arow, kb, sub);
        const __bf16* brow = wdt + (size_t)lcol * DIN;
        // B tiles hoisted in groups of 4: one load clause feeds 4 WMMAs,
        // amortizing one L2 round-trip instead of a wait per WMMA.
#pragma unroll
        for (int g = 0; g < 4; ++g) {
            v16bf b[4];
#pragma unroll
            for (int t = 0; t < 4; ++t)
                b[t] = load_a_bf16(brow + (size_t)((g * 4 + t) * 16) * DIN, kb, sub);
#pragma unroll
            for (int t = 0; t < 4; ++t)
                acc[g * 4 + t] = wmma_bf16(a, b[t], acc[g * 4 + t]);
        }
    }

    // Store fp32 z_e_down (output #2) and stage bf16 copy in LDS.
    // C/D layout: VGPR r, lanes 0-15 -> (M=r, N=lcol); lanes 16-31 -> (M=r+8).
#pragma unroll
    for (int nt = 0; nt < 16; ++nt) {
#pragma unroll
        for (int r = 0; r < 8; ++r) {
            const int m = r + sub * 8;
            const float v = acc[nt][r];
            out[O_ZED + (size_t)(rowg + m) * DEMB + nt * 16 + lcol] = v;
            lds_zd[(wave * 16 + m) * DEMB + nt * 16 + lcol] = (__bf16)v;
        }
    }
    __syncthreads();

    // Resident A fragments of this wave's z_down strip (16 x 256 bf16).
    v16bf a2[8];
    const __bf16* zrow = &lds_zd[(wave * 16 + lcol) * DEMB];
#pragma unroll
    for (int ec = 0; ec < 8; ++ec) a2[ec] = load_a_bf16(zrow, ec * 32, sub);

    // ---------------- GEMM2 + argmin over 1024 codes ----------------------
    // argmin_k ( cnorm[k] - 2 * z . c_k )   (||z||^2 is row-constant)
    float minv[8];
    int   mini[8];
#pragma unroll
    for (int r = 0; r < 8; ++r) { minv[r] = 3.4e38f; mini[r] = 0; }

    for (int blk = 0; blk < 64; ++blk) {
        const int n0 = blk * 16;
        // Prefetch the next 128B codebook line for one of the 8 K-rows
        // (cycled) -> global_prefetch_b8, hides L2 latency of later blocks.
        if (blk < 60)
            __builtin_prefetch(cbt + (size_t)(((blk & 7) * 32) + lane) * KCB + n0 + 64, 0, 3);
        // Hoist all 8 B fragments: single load clause, then 8 chained WMMAs.
        v16bf b[8];
#pragma unroll
        for (int ec = 0; ec < 8; ++ec)
            b[ec] = load_a_bf16(cbr + (size_t)(n0 + lcol) * DEMB, ec * 32, sub);
        v8f s = {};
#pragma unroll
        for (int ec = 0; ec < 8; ++ec) s = wmma_bf16(a2[ec], b[ec], s);

        const float cn = cnorm[n0 + lcol];
#pragma unroll
        for (int r = 0; r < 8; ++r) {
            const float sc = cn - 2.0f * s[r];
            if (sc < minv[r]) { minv[r] = sc; mini[r] = n0 + lcol; }
        }
    }
    // Reduce across the 16 lanes of each half (first-index tie-break).
#pragma unroll
    for (int off = 1; off < 16; off <<= 1) {
#pragma unroll
        for (int r = 0; r < 8; ++r) {
            const float ov = __shfl_xor(minv[r], off, 32);
            const int   oi = __shfl_xor(mini[r], off, 32);
            if (ov < minv[r] || (ov == minv[r] && oi < mini[r])) {
                minv[r] = ov; mini[r] = oi;
            }
        }
    }
    if (lcol == 0) {
#pragma unroll
        for (int r = 0; r < 8; ++r) {
            const int m = r + sub * 8;
            lds_code[wave * 16 + m] = mini[r];
            out[O_CODE + rowg + m] = (float)mini[r];   // output #3
        }
    }
    __syncthreads();

    // ---------------- Gather codebook rows + loss --------------------------
    const int code = lds_code[wave * 16 + lcol];       // row M = lcol
    v16bf a3[8];
    const __bf16* crow = cbr + (size_t)code * DEMB;
#pragma unroll
    for (int ec = 0; ec < 8; ++ec) a3[ec] = load_a_bf16(crow, ec * 32, sub);

    float ls = 0.0f;                                   // sum (c - z_down)^2
#pragma unroll
    for (int ec = 0; ec < 8; ++ec)
#pragma unroll
        for (int j = 0; j < 16; ++j) {
            const float d = (float)a3[ec][j] - (float)a2[ec][j];
            ls += d * d;
        }
#pragma unroll
    for (int off = 16; off > 0; off >>= 1) ls += __shfl_xor(ls, off, 32);
    if (lane == 0) lds_loss[wave] = ls;
    __syncthreads();
    if (threadIdx.x == 0) {                            // fixed-order -> deterministic
        float t = 0.0f;
#pragma unroll
        for (int w = 0; w < 8; ++w) t += lds_loss[w];
        lpart[blockIdx.x] = t;
    }

    // ---------------- GEMM3: z_q[16,512] = c[16,256] x Wu^T ----------------
    for (int nt = 0; nt < 32; ++nt) {                  // 512 output cols
        // Hoist all 8 B fragments for this output tile (one clause).
        v16bf b[8];
#pragma unroll
        for (int ec = 0; ec < 8; ++ec)
            b[ec] = load_a_bf16(wut + (size_t)(nt * 16 + lcol) * DEMB, ec * 32, sub);
        v8f a3c = {};
#pragma unroll
        for (int ec = 0; ec < 8; ++ec) a3c = wmma_bf16(a3[ec], b[ec], a3c);
#pragma unroll
        for (int r = 0; r < 8; ++r) {
            const int m = r + sub * 8;
            out[O_ZQ + (size_t)(rowg + m) * DIN + nt * 16 + lcol] = a3c[r];
        }
    }
}

// Fixed-order per-batch loss reduction (32 tiles per batch of 4096 tokens).
__global__ void vq_finalize(const float* __restrict__ lpart, float* __restrict__ out) {
    const int b = threadIdx.x;
    if (b < BB) {
        float s = 0.0f;
        for (int i = 0; i < 32; ++i) s += lpart[b * 32 + i];
        const float m = s * (1.0f / ((float)TT * (float)DEMB));
        out[O_CLOSS  + b] = m;   // commitment_loss
        out[O_CBLOSS + b] = m;   // codebook_loss (same value)
    }
}

// ---------------------------------------------------------------------------
extern "C" void kernel_launch(void* const* d_in, const int* in_sizes, int n_in,
                              void* d_out, int out_size, void* d_ws, size_t ws_size,
                              hipStream_t stream) {
    const float* z_e = (const float*)d_in[0];
    const float* cb  = (const float*)d_in[1];
    const float* Wd  = (const float*)d_in[2];
    const float* Wu  = (const float*)d_in[3];
    float* out = (float*)d_out;

    char* w = (char*)d_ws;
    void* wdt = w;            w += (size_t)DIN * DEMB * 2;   // 256 KB bf16
    void* wut = w;            w += (size_t)DEMB * DIN * 2;   // 256 KB
    void* cbt = w;            w += (size_t)DEMB * KCB * 2;   // 512 KB
    void* cbr = w;            w += (size_t)KCB * DEMB * 2;   // 512 KB
    float* cnorm = (float*)w; w += (size_t)KCB * 4;          // 4 KB
    float* lpart = (float*)w;                                // 2 KB

    vq_prep_weights <<<512, 256, 0, stream>>>(Wd, Wu, wdt, wut);
    vq_prep_codebook<<<KCB, 256, 0, stream>>>(cb, cbr, cbt, cnorm);
    vq_main<<<NTILES, 256, 0, stream>>>(z_e, wdt, wut, cbt, cbr, cnorm, lpart, out);
    vq_finalize<<<1, 32, 0, stream>>>(lpart, out);
}
